// FGHGNN_60404420051111
// MI455X (gfx1250) — matrix operand order repaired and checked by reading, not compiled, not run on hardware
//
#include <hip/hip_runtime.h>
#include <hip/hip_bf16.h>

typedef __bf16 bf16_t;
typedef __attribute__((ext_vector_type(16))) __bf16 v16bf;
typedef __attribute__((ext_vector_type(8)))  __bf16 v8bf;
typedef __attribute__((ext_vector_type(8)))  float  v8f;

#define BN_EPS 1e-5f

// ---------------------------------------------------------------------------
// Utility kernels
// ---------------------------------------------------------------------------
__global__ void k_zero_f(float* __restrict__ p, size_t n) {
    size_t i = (size_t)blockIdx.x * blockDim.x + threadIdx.x;
    if (i < n) p[i] = 0.0f;
}

// gather: out[i*128+h] = emb[idx[i]*128+h]
__global__ void k_gather128(const float* __restrict__ emb, const int* __restrict__ idx,
                            float* __restrict__ out, size_t n /* = M*128 */) {
    size_t i = (size_t)blockIdx.x * blockDim.x + threadIdx.x;
    if (i >= n) return;
    int row = idx[i >> 7];
    out[i] = emb[((size_t)row << 7) + (i & 127)];
}

// transpose + f32->bf16: Wt[n*K+k] = W[k*Nc+n]
__global__ void k_transpose_w(const float* __restrict__ W, bf16_t* __restrict__ Wt,
                              int K, int Nc) {
    int i = blockIdx.x * blockDim.x + threadIdx.x;
    if (i >= K * Nc) return;
    int k = i / Nc, n = i % Nc;
    Wt[(size_t)n * K + k] = (bf16_t)W[i];
}

// GINE message: aggr[dst] += relu(x[src] + emb[attr]); grid over E*128
__global__ void k_edge_gine(const float* __restrict__ xf, const int* __restrict__ src,
                            const int* __restrict__ dst, const int* __restrict__ attr,
                            const float* __restrict__ emb, float* __restrict__ aggr,
                            size_t tot) {
    size_t i = (size_t)blockIdx.x * blockDim.x + threadIdx.x;
    if (i >= tot) return;
    int e = (int)(i >> 7), h = (int)(i & 127);
    float m = xf[((size_t)src[e] << 7) + h] + emb[((size_t)attr[e] << 7) + h];
    m = fmaxf(m, 0.0f);
    atomicAdd(&aggr[((size_t)dst[e] << 7) + h], m);
}

// bipartite GIN message: aggr[dst] += x_src[src]
__global__ void k_edge_bip(const float* __restrict__ xf, const int* __restrict__ src,
                           const int* __restrict__ dst, float* __restrict__ aggr,
                           size_t tot) {
    size_t i = (size_t)blockIdx.x * blockDim.x + threadIdx.x;
    if (i >= tot) return;
    int e = (int)(i >> 7), h = (int)(i & 127);
    atomicAdd(&aggr[((size_t)dst[e] << 7) + h], xf[((size_t)src[e] << 7) + h]);
}

// X16 = bf16((1+eps)*x + aggr)
__global__ void k_prep(const float* __restrict__ x, const float* __restrict__ aggr,
                       const float* __restrict__ eps_ptr, int eidx,
                       bf16_t* __restrict__ out, size_t n) {
    size_t i = (size_t)blockIdx.x * blockDim.x + threadIdx.x;
    if (i >= n) return;
    float e = 1.0f + eps_ptr[eidx];
    out[i] = (bf16_t)(e * x[i] + aggr[i]);
}

// X16 = bf16(A + B)
__global__ void k_add_bf16(const float* __restrict__ A, const float* __restrict__ B,
                           bf16_t* __restrict__ out, size_t n) {
    size_t i = (size_t)blockIdx.x * blockDim.x + threadIdx.x;
    if (i >= n) return;
    out[i] = (bf16_t)(A[i] + B[i]);
}

// ---------------------------------------------------------------------------
// WMMA GEMM: Y[M,Nc] = bf16 X[M,K] @ Wt[Nc,K]^T + bias, optional relu.
// One wave -> 16x64 output tile; v_wmma_f32_16x16x32_bf16.
// ---------------------------------------------------------------------------
__global__ __launch_bounds__(256) void k_gemm_wmma(
    const bf16_t* __restrict__ X, const bf16_t* __restrict__ Wt,
    const float* __restrict__ bias, float* __restrict__ Y,
    int M, int K, int Nc, int relu, int tiles) {
    int wid = blockIdx.x * 8 + (threadIdx.x >> 5);
    if (wid >= tiles) return;
    int tilesN = Nc >> 6;
    int tm = (wid / tilesN) << 4;
    int tn = (wid % tilesN) << 6;
    int lane = threadIdx.x & 31;
    int hl   = lane >> 4;     // half-wave select
    int l16  = lane & 15;

    const bf16_t* arow = X + (size_t)(tm + l16) * K;
    v8f acc0 = {}, acc1 = {}, acc2 = {}, acc3 = {};
    const size_t bstride = (size_t)16 * K;
    const bf16_t* bcol = Wt + (size_t)(tn + l16) * K + hl * 16;

    int ksteps = K >> 5;
    for (int kk = 0; kk < ksteps; ++kk) {
        int kb = kk << 5;
        // A fragment: lane<16 rows M=l16 hold K kb+0..15; lanes>=16 hold kb+16..31
        union { v16bf v; v8bf h[2]; } a;
        a.h[0] = *(const v8bf*)(arow + kb + hl * 16);
        a.h[1] = *(const v8bf*)(arow + kb + hl * 16 + 8);
        // B fragments: lane<16 -> K rows kb+0..15 of column tn+l16 (contiguous in Wt)
        v16bf b0 = *(const v16bf*)(bcol + kb);
        v16bf b1 = *(const v16bf*)(bcol + kb + bstride);
        v16bf b2 = *(const v16bf*)(bcol + kb + 2 * bstride);
        v16bf b3 = *(const v16bf*)(bcol + kb + 3 * bstride);
        acc0 = __builtin_amdgcn_wmma_f32_16x16x32_bf16(false, a.v, false, b0, (short)0, acc0, false, false);
        acc1 = __builtin_amdgcn_wmma_f32_16x16x32_bf16(false, a.v, false, b1, (short)0, acc1, false, false);
        acc2 = __builtin_amdgcn_wmma_f32_16x16x32_bf16(false, a.v, false, b2, (short)0, acc2, false, false);
        acc3 = __builtin_amdgcn_wmma_f32_16x16x32_bf16(false, a.v, false, b3, (short)0, acc3, false, false);
    }

    int rowbase = tm + hl * 8;
    v8f accs[4] = {acc0, acc1, acc2, acc3};
#pragma unroll
    for (int nt = 0; nt < 4; ++nt) {
        int col = tn + nt * 16 + l16;
        float bv = bias[col];
        float* yp = Y + (size_t)rowbase * Nc + col;
        v8f a = accs[nt];
#pragma unroll
        for (int v = 0; v < 8; ++v) {
            float val = a[v] + bv;
            if (relu) val = fmaxf(val, 0.0f);
            yp[(size_t)v * Nc] = val;
        }
    }
}

// ---------------------------------------------------------------------------
// BatchNorm (training mode) helpers
// ---------------------------------------------------------------------------
// partial column sums over 256-row chunks; blockDim.x == C; stats[0..C)=sum, [C..2C)=sumsq
__global__ void k_bn_stats(const float* __restrict__ X, int M, int C,
                           float* __restrict__ stats) {
    int col = threadIdx.x;
    int r0 = blockIdx.x * 256;
    int r1 = r0 + 256; if (r1 > M) r1 = M;
    float s = 0.0f, s2 = 0.0f;
    for (int r = r0; r < r1; ++r) {
        float v = X[(size_t)r * C + col];
        s += v; s2 += v * v;
    }
    atomicAdd(&stats[col], s);
    atomicAdd(&stats[C + col], s2);
}

__global__ void k_bn_apply(float* __restrict__ X, int M, int C,
                           const float* __restrict__ stats,
                           const float* __restrict__ g, const float* __restrict__ b,
                           int relu) {
    size_t i = (size_t)blockIdx.x * blockDim.x + threadIdx.x;
    if (i >= (size_t)M * C) return;
    int c = (int)(i % C);
    float mu = stats[c] / (float)M;
    float var = stats[C + c] / (float)M - mu * mu;
    float v = (X[i] - mu) * rsqrtf(var + BN_EPS) * g[c] + b[c];
    if (relu) v = fmaxf(v, 0.0f);
    X[i] = v;
}

// xf[i] += maybe_relu(bn(D[i]))
__global__ void k_bn_residual(const float* __restrict__ D, float* __restrict__ xf,
                              int M, int C, const float* __restrict__ stats,
                              const float* __restrict__ g, const float* __restrict__ b,
                              int relu) {
    size_t i = (size_t)blockIdx.x * blockDim.x + threadIdx.x;
    if (i >= (size_t)M * C) return;
    int c = (int)(i % C);
    float mu = stats[c] / (float)M;
    float var = stats[C + c] / (float)M - mu * mu;
    float v = (D[i] - mu) * rsqrtf(var + BN_EPS) * g[c] + b[c];
    if (relu) v = fmaxf(v, 0.0f);
    xf[i] += v;
}

// ---------------------------------------------------------------------------
// Pooling & classifier
// ---------------------------------------------------------------------------
__global__ void k_pool_sum(const float* __restrict__ x, const int* __restrict__ batch,
                           float* __restrict__ pool, size_t tot) {
    size_t i = (size_t)blockIdx.x * blockDim.x + threadIdx.x;
    if (i >= tot) return;
    atomicAdd(&pool[((size_t)batch[i >> 7] << 7) + (i & 127)], x[i]);
}

__global__ void k_count(const int* __restrict__ batch, float* __restrict__ cnt, int M) {
    int i = blockIdx.x * blockDim.x + threadIdx.x;
    if (i >= M) return;
    atomicAdd(&cnt[batch[i]], 1.0f);
}

__global__ void k_pool_combine(const float* __restrict__ pa, const float* __restrict__ ca,
                               const float* __restrict__ pc, const float* __restrict__ cc,
                               bf16_t* __restrict__ out, size_t tot) {
    size_t i = (size_t)blockIdx.x * blockDim.x + threadIdx.x;
    if (i >= tot) return;
    size_t gidx = i >> 7;
    float va = pa[i] / fmaxf(ca[gidx], 1.0f);
    float vc = pc[i] / fmaxf(cc[gidx], 1.0f);
    out[i] = (bf16_t)(va + vc);
}

// out[g] = dot(hidden[g,0:256], W2) + b2
__global__ void k_cls_out(const float* __restrict__ hidden, const float* __restrict__ W2,
                          const float* __restrict__ b2, float* __restrict__ out, int Gn) {
    int gidx = blockIdx.x * blockDim.x + threadIdx.x;
    if (gidx >= Gn) return;
    float s = b2[0];
    const float* h = hidden + (size_t)gidx * 256;
    for (int j = 0; j < 256; ++j) s += h[j] * W2[j];
    out[gidx] = s;
}

// ---------------------------------------------------------------------------
// Host orchestration
// ---------------------------------------------------------------------------
extern "C" void kernel_launch(void* const* d_in, const int* in_sizes, int n_in,
                              void* d_out, int out_size, void* d_ws, size_t ws_size,
                              hipStream_t stream) {
    const int N = 200000, NC = 50000, E = 500000, EC = 150000, EB = 200000;
    const int H = 128, P = 256, L = 4, G = 8192;
    const int WEL = H * P; // 32768 elements per weight slab

    // inputs (setup_inputs dict order)
    const int*   x_idx      = (const int*)d_in[0];
    const int*   xc_idx     = (const int*)d_in[1];
    const int*   edge_index = (const int*)d_in[2];
    const int*   edge_attr  = (const int*)d_in[3];
    const int*   c2c_ei     = (const int*)d_in[4];
    const int*   c2c_attr   = (const int*)d_in[5];
    const int*   a2c_ei     = (const int*)d_in[6];
    const int*   c2a_ei     = (const int*)d_in[7];
    const int*   x_batch    = (const int*)d_in[8];
    const int*   xc_batch   = (const int*)d_in[9];
    const float* atom_emb   = (const float*)d_in[10];
    const float* clus_emb   = (const float*)d_in[11];
    const float* bond_emb   = (const float*)d_in[12];
    const float* c2c_emb    = (const float*)d_in[13];
    const float* ag_eps = (const float*)d_in[14]; const float* ag_W = (const float*)d_in[15];
    const float* ag_b  = (const float*)d_in[16];  const float* ag_g = (const float*)d_in[17];
    const float* ag_be = (const float*)d_in[18];
    const float* cg_eps = (const float*)d_in[19]; const float* cg_W = (const float*)d_in[20];
    const float* cg_b  = (const float*)d_in[21];  const float* cg_g = (const float*)d_in[22];
    const float* cg_be = (const float*)d_in[23];
    const float* a2c_eps = (const float*)d_in[24]; const float* a2c_W = (const float*)d_in[25];
    const float* a2c_b = (const float*)d_in[26];   const float* a2c_g = (const float*)d_in[27];
    const float* a2c_be = (const float*)d_in[28];
    const float* c2a_eps = (const float*)d_in[29]; const float* c2a_W = (const float*)d_in[30];
    const float* c2a_b = (const float*)d_in[31];   const float* c2a_g = (const float*)d_in[32];
    const float* c2a_be = (const float*)d_in[33];
    const float* ma_W = (const float*)d_in[34]; const float* ma_b = (const float*)d_in[35];
    const float* mc_W = (const float*)d_in[36]; const float* mc_b = (const float*)d_in[37];
    const float* abn_g = (const float*)d_in[38]; const float* abn_b = (const float*)d_in[39];
    const float* cbn_g = (const float*)d_in[40]; const float* cbn_b = (const float*)d_in[41];
    const float* cl_W1 = (const float*)d_in[42]; const float* cl_b1 = (const float*)d_in[43];
    const float* cl_W2 = (const float*)d_in[44]; const float* cl_b2 = (const float*)d_in[45];

    // workspace layout (256-byte aligned slabs)
    char* ws = (char*)d_ws;
    size_t off = 0;
    auto take = [&](size_t bytes) -> void* {
        void* p = ws + off;
        off += (bytes + 255) & ~(size_t)255;
        return p;
    };
    float*  xa    = (float*)take((size_t)N * H * 4);
    float*  xc    = (float*)take((size_t)NC * H * 4);
    float*  aggrA = (float*)take((size_t)N * H * 4);
    float*  aggrC = (float*)take((size_t)NC * H * 4);
    float*  ZA    = (float*)take((size_t)N * P * 4);
    float*  ZB    = (float*)take((size_t)N * P * 4);
    float*  ZC    = (float*)take((size_t)NC * P * 4);
    float*  ZD    = (float*)take((size_t)NC * P * 4);
    bf16_t* X16   = (bf16_t*)take((size_t)N * P * 2);
    bf16_t* W16   = (bf16_t*)take((size_t)19 * WEL * 2);
    float*  stats = (float*)take(2 * P * 4);
    float*  poolA = (float*)take((size_t)G * H * 4);
    float*  poolC = (float*)take((size_t)G * H * 4);
    float*  cntA  = (float*)take((size_t)G * 4);
    float*  cntC  = (float*)take((size_t)G * 4);
    float*  hidden= (float*)take((size_t)G * P * 4);

    bf16_t* agW16  = W16;
    bf16_t* cgW16  = W16 + 4 * WEL;
    bf16_t* a2cW16 = W16 + 8 * WEL;
    bf16_t* c2aW16 = W16 + 9 * WEL;
    bf16_t* maW16  = W16 + 10 * WEL;
    bf16_t* mcW16  = W16 + 14 * WEL;
    bf16_t* clW16  = W16 + 18 * WEL;

    auto gridFor = [](size_t n) { return (unsigned)((n + 255) / 256); };
    auto zf = [&](float* p, size_t n) {
        k_zero_f<<<gridFor(n), 256, 0, stream>>>(p, n);
    };
    auto transp = [&](const float* W, bf16_t* Wt, int K, int Nc) {
        k_transpose_w<<<gridFor((size_t)K * Nc), 256, 0, stream>>>(W, Wt, K, Nc);
    };
    auto gemm = [&](const bf16_t* Xb, const bf16_t* Wt, const float* bias, float* Y,
                    int M, int K, int Nc, int relu) {
        int tiles = (M / 16) * (Nc / 64);
        k_gemm_wmma<<<(tiles + 7) / 8, 256, 0, stream>>>(Xb, Wt, bias, Y, M, K, Nc, relu, tiles);
    };

    // --- weights -> transposed bf16 ---
    for (int l = 0; l < L; ++l) {
        transp(ag_W + (size_t)l * WEL, agW16 + (size_t)l * WEL, H, P);
        transp(cg_W + (size_t)l * WEL, cgW16 + (size_t)l * WEL, H, P);
        transp(ma_W + (size_t)l * WEL, maW16 + (size_t)l * WEL, P, H);
        transp(mc_W + (size_t)l * WEL, mcW16 + (size_t)l * WEL, P, H);
    }
    transp(a2c_W, a2cW16, H, P);
    transp(c2a_W, c2aW16, H, P);
    transp(cl_W1, clW16, H, P);

    // --- initial embeddings ---
    k_gather128<<<gridFor((size_t)N * H), 256, 0, stream>>>(atom_emb, x_idx, xa, (size_t)N * H);
    k_gather128<<<gridFor((size_t)NC * H), 256, 0, stream>>>(clus_emb, xc_idx, xc, (size_t)NC * H);

    // --- GINE conv (mlp output [M,256] into Z) ---
    auto run_gine = [&](const float* xf, int M, const int* ei, int Eg,
                        const int* eattr, const float* eemb,
                        const float* eps_ptr, int eidx,
                        const bf16_t* Wt, const float* bias,
                        const float* g, const float* be,
                        float* aggr, float* Z) {
        zf(aggr, (size_t)M * H);
        size_t tot = (size_t)Eg * H;
        k_edge_gine<<<gridFor(tot), 256, 0, stream>>>(xf, ei, ei + Eg, eattr, eemb, aggr, tot);
        k_prep<<<gridFor((size_t)M * H), 256, 0, stream>>>(xf, aggr, eps_ptr, eidx, X16, (size_t)M * H);
        gemm(X16, Wt, bias, Z, M, H, P, 0);
        zf(stats, 2 * P);
        k_bn_stats<<<(M + 255) / 256, P, 0, stream>>>(Z, M, P, stats);
        k_bn_apply<<<gridFor((size_t)M * P), 256, 0, stream>>>(Z, M, P, stats, g, be, 1);
    };

    // --- bipartite GIN conv ---
    auto run_bip = [&](const float* xsrc, const float* xdst, int Mdst, const int* ei, int Eg,
                       const float* eps_ptr, int eidx,
                       const bf16_t* Wt, const float* bias,
                       const float* g, const float* be,
                       float* aggr, float* Z) {
        zf(aggr, (size_t)Mdst * H);
        size_t tot = (size_t)Eg * H;
        k_edge_bip<<<gridFor(tot), 256, 0, stream>>>(xsrc, ei, ei + Eg, aggr, tot);
        k_prep<<<gridFor((size_t)Mdst * H), 256, 0, stream>>>(xdst, aggr, eps_ptr, eidx, X16, (size_t)Mdst * H);
        gemm(X16, Wt, bias, Z, Mdst, H, P, 0);
        zf(stats, 2 * P);
        k_bn_stats<<<(Mdst + 255) / 256, P, 0, stream>>>(Z, Mdst, P, stats);
        k_bn_apply<<<gridFor((size_t)Mdst * P), 256, 0, stream>>>(Z, Mdst, P, stats, g, be, 1);
    };

    // --- combine two [M,256] branches, project to H, BN, (relu), residual ---
    auto stage2 = [&](const float* Z1, const float* Z2, int M,
                      const bf16_t* Wt, const float* bias,
                      const float* g, const float* b, int relu,
                      float* D, float* xf) {
        size_t n2 = (size_t)M * P;
        k_add_bf16<<<gridFor(n2), 256, 0, stream>>>(Z1, Z2, X16, n2);
        gemm(X16, Wt, bias, D, M, P, H, 0);
        zf(stats, 2 * H);
        k_bn_stats<<<(M + 255) / 256, H, 0, stream>>>(D, M, H, stats);
        size_t n1 = (size_t)M * H;
        k_bn_residual<<<gridFor(n1), 256, 0, stream>>>(D, xf, M, H, stats, g, b, relu);
    };

    // --- layers ---
    for (int l = 0; l < L; ++l) {
        run_gine(xa, N, edge_index, E, edge_attr, bond_emb, ag_eps, l,
                 agW16 + (size_t)l * WEL, ag_b + (size_t)l * P,
                 ag_g + (size_t)l * P, ag_be + (size_t)l * P, aggrA, ZA);
        run_bip(xc, xa, N, c2a_ei, EB, c2a_eps, 0, c2aW16, c2a_b, c2a_g, c2a_be, aggrA, ZB);
        run_gine(xc, NC, c2c_ei, EC, c2c_attr, c2c_emb, cg_eps, l,
                 cgW16 + (size_t)l * WEL, cg_b + (size_t)l * P,
                 cg_g + (size_t)l * P, cg_be + (size_t)l * P, aggrC, ZC);
        run_bip(xa, xc, NC, a2c_ei, EB, a2c_eps, 0, a2cW16, a2c_b, a2c_g, a2c_be, aggrC, ZD);
        int relu = (l < L - 1) ? 1 : 0;
        stage2(ZA, ZB, N, maW16 + (size_t)l * WEL, ma_b + (size_t)l * H,
               abn_g + (size_t)l * H, abn_b + (size_t)l * H, relu, aggrA, xa);
        stage2(ZC, ZD, NC, mcW16 + (size_t)l * WEL, mc_b + (size_t)l * H,
               cbn_g + (size_t)l * H, cbn_b + (size_t)l * H, relu, aggrC, xc);
    }

    // --- mean pooling ---
    zf(poolA, (size_t)G * H); zf(poolC, (size_t)G * H);
    zf(cntA, G); zf(cntC, G);
    k_pool_sum<<<gridFor((size_t)N * H), 256, 0, stream>>>(xa, x_batch, poolA, (size_t)N * H);
    k_pool_sum<<<gridFor((size_t)NC * H), 256, 0, stream>>>(xc, xc_batch, poolC, (size_t)NC * H);
    k_count<<<(N + 255) / 256, 256, 0, stream>>>(x_batch, cntA, N);
    k_count<<<(NC + 255) / 256, 256, 0, stream>>>(xc_batch, cntC, NC);
    k_pool_combine<<<gridFor((size_t)G * H), 256, 0, stream>>>(poolA, cntA, poolC, cntC, X16, (size_t)G * H);

    // --- classifier ---
    gemm(X16, clW16, cl_b1, hidden, G, H, P, 1);
    k_cls_out<<<(G + 255) / 256, 256, 0, stream>>>(hidden, cl_W2, cl_b2, (float*)d_out, G);
}
